// VectorQuantizer_12592844112179
// MI455X (gfx1250) — matrix operand order repaired and checked by reading, not compiled, not run on hardware
//
#include <hip/hip_runtime.h>
#include <math.h>

#define DIM    256
#define KCODES 1024
#define NROWS  65536
#define TILE_CODES 32
#define BSTRIDE 264     // LDS row stride in bf16 units: dword stride 132 %64==4 -> conflict-free; byte stride 528 %16==0

typedef float  v8f   __attribute__((ext_vector_type(8)));
typedef __bf16 v8bf  __attribute__((ext_vector_type(8)));
typedef __bf16 v16bf __attribute__((ext_vector_type(16)));

// ---------------------------------------------------------------------------
// Kernel 1: split codebook into bf16 hi/lo planes (x ~= hi + lo, err ~2^-17)
// ---------------------------------------------------------------------------
__global__ void vq_split_kernel(const float* __restrict__ cb,
                                __bf16* __restrict__ cbh,
                                __bf16* __restrict__ cbl) {
    int i = blockIdx.x * 256 + threadIdx.x;      // KCODES*DIM / 256 blocks
    float v = cb[i];
    __bf16 h = (__bf16)v;
    cbh[i] = h;
    cbl[i] = (__bf16)(v - (float)h);
}

// ---------------------------------------------------------------------------
// Kernel 2: per-code squared norms (exact f32) + zero the histogram.
// ---------------------------------------------------------------------------
__global__ void vq_norms_kernel(const float* __restrict__ cb,
                                float* __restrict__ norm2,
                                unsigned* __restrict__ counts) {
    int k    = blockIdx.x;
    int lane = threadIdx.x;            // 32 threads
    const float* row = cb + (size_t)k * DIM;
    float s = 0.0f;
    #pragma unroll
    for (int i = 0; i < DIM / 32; ++i) {
        float v = row[lane + i * 32];
        s += v * v;
    }
    #pragma unroll
    for (int m = 16; m >= 1; m >>= 1) s += __shfl_xor(s, m, 32);
    if (lane == 0) { norm2[k] = s; counts[k] = 0u; }
}

// ---------------------------------------------------------------------------
// Kernel 3: fused score-GEMM (bf16-split WMMA, f32 accum) + argmin + gather
//           + histogram. 4 waves/block, each wave owns a 16-row tile.
//           score = x.e ~= xh.eh + xh.el + xl.eh  (3 chained 16x16x32 WMMAs/kb)
// ---------------------------------------------------------------------------
__global__ void __launch_bounds__(128) vq_main_kernel(
        const float*  __restrict__ ze,
        const float*  __restrict__ cb,
        const __bf16* __restrict__ cbh,
        const __bf16* __restrict__ cbl,
        const float*  __restrict__ norm2,
        float* __restrict__ zq,
        unsigned* __restrict__ counts) {
    __shared__ __bf16 lds_hi[TILE_CODES * BSTRIDE];   // 16,896 B
    __shared__ __bf16 lds_lo[TILE_CODES * BSTRIDE];   // 16,896 B

    const int tid  = threadIdx.x;
    const int wave = tid >> 5;
    const int lane = tid & 31;
    const int hsel = lane >> 4;        // half-wave select
    const int lrow = lane & 15;        // M for A-frag, N (code column) for B/C
    const int rowbase = blockIdx.x * 64 + wave * 16;

    // ---- Build the wave's A tile as bf16 hi/lo fragments in VGPRs (once) ----
    // 16-bit A 16x32 layout: lane<16 holds K = {0..7, 16..23}; lane>=16 holds
    // K = {8..15, 24..31}; elements 0..7 -> first run, 8..15 -> second run.
    v16bf ahi[8], alo[8];
    {
        const float* arow = ze + (size_t)(rowbase + lrow) * DIM + 8 * hsel;
        #pragma unroll
        for (int kb = 0; kb < 8; ++kb) {
            float vals[16];
            const float* p = arow + kb * 32;
            *(float4*)&vals[0]  = *(const float4*)(p);
            *(float4*)&vals[4]  = *(const float4*)(p + 4);
            *(float4*)&vals[8]  = *(const float4*)(p + 16);
            *(float4*)&vals[12] = *(const float4*)(p + 20);
            v16bf h, l;
            #pragma unroll
            for (int e = 0; e < 16; ++e) {
                float  f  = vals[e];
                __bf16 hh = (__bf16)f;
                h[e] = hh;
                l[e] = (__bf16)(f - (float)hh);
            }
            ahi[kb] = h;
            alo[kb] = l;
        }
    }

    float bestv[8];
    int   besti[8];
    #pragma unroll
    for (int v = 0; v < 8; ++v) { bestv[v] = 3.4e38f; besti[v] = 0; }

    for (int kt = 0; kt < KCODES / TILE_CODES; ++kt) {
        // ---- Cooperative stage of 32 bf16 codebook rows (hi+lo) into LDS ----
        {
            int r = tid >> 2, part = tid & 3;          // 4 threads/row, 64 bf16 each
            size_t g = (size_t)(kt * TILE_CODES + r) * DIM + part * 64;
            int    o = r * BSTRIDE + part * 64;
            const float4* sh = (const float4*)(cbh + g);   // 8 x 16B per plane
            const float4* sl = (const float4*)(cbl + g);
            float4* dh = (float4*)(lds_hi + o);
            float4* dl = (float4*)(lds_lo + o);
            #pragma unroll
            for (int j = 0; j < 8; ++j) { dh[j] = sh[j]; dl[j] = sl[j]; }
        }
        __syncthreads();

        // ---- Two 16-code column sub-tiles ----
        #pragma unroll
        for (int t = 0; t < 2; ++t) {
            v8f c = {};
            const int rowoff = (t * 16 + lrow) * BSTRIDE + 8 * hsel;
            #pragma unroll
            for (int kb = 0; kb < 8; ++kb) {
                const __bf16* ph = lds_hi + rowoff + kb * 32;
                const __bf16* pl = lds_lo + rowoff + kb * 32;
                v8bf h0 = *(const v8bf*)(ph);
                v8bf h1 = *(const v8bf*)(ph + 16);
                v8bf l0 = *(const v8bf*)(pl);
                v8bf l1 = *(const v8bf*)(pl + 16);
                v16bf bh = __builtin_shufflevector(h0, h1, 0,1,2,3,4,5,6,7,8,9,10,11,12,13,14,15);
                v16bf bl = __builtin_shufflevector(l0, l1, 0,1,2,3,4,5,6,7,8,9,10,11,12,13,14,15);
                c = __builtin_amdgcn_wmma_f32_16x16x32_bf16(false, ahi[kb], false, bh,
                                                            (short)0, c, false, false);
                c = __builtin_amdgcn_wmma_f32_16x16x32_bf16(false, alo[kb], false, bh,
                                                            (short)0, c, false, false);
                c = __builtin_amdgcn_wmma_f32_16x16x32_bf16(false, ahi[kb], false, bl,
                                                            (short)0, c, false, false);
            }
            // Lane holds 8 scores: row = v + 8*hsel, column code = base + lrow.
            const int   code = kt * TILE_CODES + t * 16 + lrow;
            const float nrm  = norm2[code];
            #pragma unroll
            for (int v = 0; v < 8; ++v) {
                float dist   = nrm - 2.0f * c[v];
                bool  better = dist < bestv[v];        // strict: ties keep lower code
                besti[v] = better ? code : besti[v];
                bestv[v] = better ? dist : bestv[v];
            }
        }
        __syncthreads();
    }

    // ---- Cross-column argmin: butterfly over the 16 lanes of each half-wave ----
    #pragma unroll
    for (int v = 0; v < 8; ++v) {
        float bv = bestv[v];
        int   bi = besti[v];
        #pragma unroll
        for (int m = 8; m >= 1; m >>= 1) {
            float ov = __shfl_xor(bv, m, 32);
            int   oi = __shfl_xor(bi, m, 32);
            if (ov < bv || (ov == bv && oi < bi)) { bv = ov; bi = oi; }
        }
        bestv[v] = bv; besti[v] = bi;
    }

    // ---- Gather z_q rows (exact f32 codebook) + histogram ----
    #pragma unroll
    for (int r = 0; r < 16; ++r) {
        int idx = __shfl(besti[r & 7], (r < 8) ? 0 : 16, 32);
        const float4* src = (const float4*)(cb + (size_t)idx * DIM);
        float4*       dst = (float4*)(zq + (size_t)(rowbase + r) * DIM);
        dst[lane]      = src[lane];
        dst[lane + 32] = src[lane + 32];
        if (lane == 0) atomicAdd(&counts[idx], 1u);
    }
}

// ---------------------------------------------------------------------------
// Kernel 4: perplexity = exp(-sum p*log(p+1e-10)), p = counts/N
// ---------------------------------------------------------------------------
__global__ void vq_perplexity_kernel(const unsigned* __restrict__ counts,
                                     float* __restrict__ out) {
    __shared__ float red[KCODES];
    int t = threadIdx.x;
    float p = (float)counts[t] * (1.0f / (float)NROWS);
    red[t] = p * logf(p + 1e-10f);
    __syncthreads();
    for (int s = KCODES / 2; s >= 1; s >>= 1) {
        if (t < s) red[t] += red[t + s];
        __syncthreads();
    }
    if (t == 0) out[(size_t)NROWS * DIM] = expf(-red[0]);
}

// ---------------------------------------------------------------------------
extern "C" void kernel_launch(void* const* d_in, const int* in_sizes, int n_in,
                              void* d_out, int out_size, void* d_ws, size_t ws_size,
                              hipStream_t stream) {
    const float* ze = (const float*)d_in[0];   // [16,64,64,256] f32
    const float* cb = (const float*)d_in[1];   // [1024,256] f32
    float* out = (float*)d_out;                // z_q flat + perplexity at the end

    // ws layout: norm2 (4KB) | counts (4KB) | cb_hi (512KB) | cb_lo (512KB)
    float*    norm2  = (float*)d_ws;
    unsigned* counts = (unsigned*)((char*)d_ws + KCODES * sizeof(float));
    __bf16*   cbh    = (__bf16*)((char*)d_ws + 2 * KCODES * sizeof(float));
    __bf16*   cbl    = cbh + (size_t)KCODES * DIM;

    vq_split_kernel<<<(KCODES * DIM) / 256, 256, 0, stream>>>(cb, cbh, cbl);
    vq_norms_kernel<<<KCODES, 32, 0, stream>>>(cb, norm2, counts);
    vq_main_kernel<<<NROWS / 64, 128, 0, stream>>>(ze, cb, cbh, cbl, norm2, out, counts);
    vq_perplexity_kernel<<<1, KCODES, 0, stream>>>(counts, out);
}
